// HybridDynamicsModel_69518340653125
// MI455X (gfx1250) — matrix operand order repaired and checked by reading, not compiled, not run on hardware
//
#include <hip/hip_runtime.h>

typedef _Float16 v16h __attribute__((ext_vector_type(16)));
typedef float    v8f  __attribute__((ext_vector_type(8)));

#define IN_DIM 160
#define HID    64
#define NX     32
#define NMODES 8

// fragment-index bases (units of 512 f16 = one 32(K) x 16(N) B-tile)
#define FB_C1 0      // Wc1: 5 K-tiles x 4 N-tiles = 20 frags
#define FB_C2 20     // Wc2: 2 x 4 = 8
#define FB_C3 28     // Wc3: 2 x 4 = 8
#define FB_C4 36     // Wc4: 2 x 1 (N padded 8->16) = 2
#define FB_E1 38     // We1: 8 experts x 5 x 4 = 160
#define FB_E2 198    // We2: 8 x 2 x 4 = 64
#define FB_E3 262    // We3: 8 x 2 x 4 = 64
#define FB_E4 326    // We4: 8 x 2 x 2 = 32
#define FB_TOTAL 358 // ~367 KB of d_ws as f16

// ---------------------------------------------------------------------------
// Pack fp32 weights [E][K][N] into f16 WMMA B-fragments.
// B layout (ISA 7.12.2, 16-bit B 32x16): n = lane&15 ; k = (e&7) + 16*(e>=8) + 8*(lane>=16)
// Fragment stored [frag][lane][16] so each lane streams 32 contiguous bytes.
// ---------------------------------------------------------------------------
__global__ __launch_bounds__(512) void pack_frags(const float* __restrict__ src,
                                                  _Float16* __restrict__ dst,
                                                  int K, int N, int KT, int NT) {
  int t    = threadIdx.x;        // 0..511 = one fragment element
  int lane = t >> 4;
  int e    = t & 15;
  int fid  = blockIdx.x;         // global fragment id (expert folded in)
  int fpe  = KT * NT;
  int ex   = fid / fpe;
  int rem  = fid - ex * fpe;
  int kt   = rem / NT;
  int nt   = rem - kt * NT;
  int k = kt * 32 + (e & 7) + ((e >> 3) << 4) + ((lane >> 4) << 3);
  int n = nt * 16 + (lane & 15);
  float v = (k < K && n < N) ? src[(size_t)ex * K * N + (size_t)k * N + n] : 0.0f;
  dst[(size_t)fid * 512 + (size_t)lane * 16 + e] = (_Float16)v;
}

// ---------------------------------------------------------------------------
// Device helpers
// ---------------------------------------------------------------------------
__device__ __forceinline__ v16h load_bfrag(const _Float16* wp, int fid, int lane) {
  return *(const v16h*)(wp + (size_t)fid * 512 + (size_t)lane * 16);
}

__device__ __forceinline__ v8f wmma_f16(v16h a, v16h b, v8f c) {
  // (neg_a, A, neg_b, B, c_mod, C, reuse_a, reuse_b)
  return __builtin_amdgcn_wmma_f32_16x16x32_f16(false, a, false, b, (short)0, c,
                                                false, false);
}

// D tile (32-bit C/D 16x16): vgpr r -> row m = r + 8*(lane>=16), col n = lane&15.
// Scatter bias+ReLU result into LDS laid out as f16 A-fragments:
// A layout: element (lane', e') = row (lane'&15),
//           k = (e'&7) + 16*(e'>=8) + 8*(lane'>=16)
__device__ __forceinline__ void store_act_f16(_Float16* afr, int mt, v8f c,
                                              const float* bias, int nt,
                                              int lane) {
  int n   = lane & 15;
  int hi  = lane >> 4;
  int k   = nt * 16 + n;                 // global hidden index 0..63
  int kt  = k >> 5;
  int kk  = k & 31;
  int hi2 = (kk >> 3) & 1;
  int e   = (kk & 7) | ((kk >> 4) << 3);
  float bv = bias[k];
  _Float16* base = afr + (mt * 2 + kt) * 512;
#pragma unroll
  for (int r = 0; r < 8; ++r) {
    float v = fmaxf(c[r] + bv, 0.0f);
    int lane2 = (r + 8 * hi) + (hi2 << 4);
    base[lane2 * 16 + e] = (_Float16)v;
  }
}

// Reload an A-fragment: single contiguous 32B per lane (2x ds_load_b128)
__device__ __forceinline__ v16h load_afrag(const _Float16* afr, int mt, int kt,
                                           int lane) {
  return *(const v16h*)(afr + (mt * 2 + kt) * 512 + lane * 16);
}

// One 64->64 hidden layer for both M-tiles: B fragment loaded once, used twice.
__device__ __forceinline__ void hidden_layer(v16h ah[2][2], const _Float16* wp,
                                             int fbase, _Float16* afr,
                                             const float* bias, int lane) {
#pragma unroll
  for (int nt = 0; nt < 4; ++nt) {
    v8f c0 = {}, c1 = {};
#pragma unroll
    for (int kt = 0; kt < 2; ++kt) {
      v16h b = load_bfrag(wp, fbase + kt * 4 + nt, lane);
      c0 = wmma_f16(ah[0][kt], b, c0);
      c1 = wmma_f16(ah[1][kt], b, c1);
    }
    store_act_f16(afr, 0, c0, bias, nt, lane);
    store_act_f16(afr, 1, c1, bias, nt, lane);
  }
  __syncthreads();
#pragma unroll
  for (int mt = 0; mt < 2; ++mt) {
    ah[mt][0] = load_afrag(afr, mt, 0, lane);
    ah[mt][1] = load_afrag(afr, mt, 1, lane);
  }
}

// ---------------------------------------------------------------------------
// Main fused kernel: classifier + argmax routing + dense expert sweep with
// routed scatter. 4 waves/block, 32 tokens/wave (two 16-row M-tiles).
// ---------------------------------------------------------------------------
__global__ __launch_bounds__(128) void moe_kernel(
    const float* __restrict__ obs, const _Float16* __restrict__ wp,
    const float* __restrict__ bc1, const float* __restrict__ bc2,
    const float* __restrict__ bc3, const float* __restrict__ bc4,
    const float* __restrict__ be1, const float* __restrict__ be2,
    const float* __restrict__ be3, const float* __restrict__ be4,
    float* __restrict__ out) {
  // per-wave f16 activation staging in A-fragment layout: [mtile][ktile][512]
  __shared__ __align__(16) _Float16 s_afrag[4][2][2][512];  // 16 KB
  __shared__ float s_logit[4][32 * NMODES];                 // 4 KB
  __shared__ int   s_mode[4][32];

  const int lane = threadIdx.x & 31;
  const int wid  = threadIdx.x >> 5;
  const int m    = lane & 15;
  const int hi   = lane >> 4;
  const int n    = m;  // B/C/D column index for this lane
  const size_t tokBase = ((size_t)blockIdx.x * 4 + wid) * 32;

  _Float16* afr = &s_afrag[wid][0][0][0];
  float*    lg  = s_logit[wid];
  int*      md  = s_mode[wid];

  // ---- obs -> f16 A-fragments for both M-tiles (kept live throughout) ----
  v16h a0[2][5];
#pragma unroll
  for (int mt = 0; mt < 2; ++mt) {
    const float* orow = obs + (tokBase + mt * 16 + m) * IN_DIM;
#pragma unroll
    for (int kt = 0; kt < 5; ++kt) {
      v16h a;
#pragma unroll
      for (int p = 0; p < 8; ++p) {
        int e = 2 * p;
        int k = kt * 32 + (e & 7) + ((e >> 3) << 4) + (hi << 3);
        float2 v = *(const float2*)(orow + k);
        a[e]     = (_Float16)v.x;
        a[e + 1] = (_Float16)v.y;
      }
      a0[mt][kt] = a;
    }
  }

  v16h ah[2][2];

  // =============== mode classifier ===============
  // L1: 160 -> 64
#pragma unroll
  for (int nt = 0; nt < 4; ++nt) {
    v8f c0 = {}, c1 = {};
#pragma unroll
    for (int kt = 0; kt < 5; ++kt) {
      v16h b = load_bfrag(wp, FB_C1 + kt * 4 + nt, lane);
      c0 = wmma_f16(a0[0][kt], b, c0);
      c1 = wmma_f16(a0[1][kt], b, c1);
    }
    store_act_f16(afr, 0, c0, bc1, nt, lane);
    store_act_f16(afr, 1, c1, bc1, nt, lane);
  }
  __syncthreads();
#pragma unroll
  for (int mt = 0; mt < 2; ++mt) {
    ah[mt][0] = load_afrag(afr, mt, 0, lane);
    ah[mt][1] = load_afrag(afr, mt, 1, lane);
  }

  hidden_layer(ah, wp, FB_C2, afr, bc2, lane);  // L2
  hidden_layer(ah, wp, FB_C3, afr, bc3, lane);  // L3

  // L4: 64 -> 8 (N padded to 16 with zero weights)
  {
    v16h b0 = load_bfrag(wp, FB_C4 + 0, lane);
    v16h b1 = load_bfrag(wp, FB_C4 + 1, lane);
#pragma unroll
    for (int mt = 0; mt < 2; ++mt) {
      v8f c = {};
      c = wmma_f16(ah[mt][0], b0, c);
      c = wmma_f16(ah[mt][1], b1, c);
      if (n < NMODES) {
        float bv = bc4[n];
#pragma unroll
        for (int r = 0; r < 8; ++r)
          lg[(mt * 16 + r + 8 * hi) * NMODES + n] = c[r] + bv;
      }
    }
  }
  __syncthreads();

  // per-token argmax (first-max wins, matching jnp.argmax); lane = token
  {
    float best = lg[lane * NMODES + 0];
    int bi = 0;
#pragma unroll
    for (int j = 1; j < NMODES; ++j) {
      float v = lg[lane * NMODES + j];
      if (v > best) { best = v; bi = j; }
    }
    md[lane] = bi;
  }
  __syncthreads();

  int mode_r[2][8];
#pragma unroll
  for (int mt = 0; mt < 2; ++mt)
#pragma unroll
    for (int r = 0; r < 8; ++r) mode_r[mt][r] = md[mt * 16 + r + 8 * hi];

  // =============== dense expert sweep with routed store ===============
  for (int ex = 0; ex < NMODES; ++ex) {
    // prefetch next expert's L1 fragments into cache (global_prefetch_b8)
    {
      const char* pf =
          (const char*)(wp + (size_t)(FB_E1 + ((ex + 1) & 7) * 20) * 512);
#pragma unroll
      for (int i = 0; i < 5; ++i)
        __builtin_prefetch(pf + ((size_t)i * 32 + lane) * 128, 0, 3);
    }

    // L1: 160 -> 64 (reuses obs fragments; B fragment shared by both M-tiles)
#pragma unroll
    for (int nt = 0; nt < 4; ++nt) {
      v8f c0 = {}, c1 = {};
#pragma unroll
      for (int kt = 0; kt < 5; ++kt) {
        v16h b = load_bfrag(wp, FB_E1 + (ex * 5 + kt) * 4 + nt, lane);
        c0 = wmma_f16(a0[0][kt], b, c0);
        c1 = wmma_f16(a0[1][kt], b, c1);
      }
      store_act_f16(afr, 0, c0, be1 + ex * HID, nt, lane);
      store_act_f16(afr, 1, c1, be1 + ex * HID, nt, lane);
    }
    __syncthreads();
#pragma unroll
    for (int mt = 0; mt < 2; ++mt) {
      ah[mt][0] = load_afrag(afr, mt, 0, lane);
      ah[mt][1] = load_afrag(afr, mt, 1, lane);
    }

    hidden_layer(ah, wp, FB_E2 + ex * 8, afr, be2 + ex * HID, lane);  // L2
    hidden_layer(ah, wp, FB_E3 + ex * 8, afr, be3 + ex * HID, lane);  // L3

    // L4: 64 -> 32, scatter rows routed to this expert
#pragma unroll
    for (int nt = 0; nt < 2; ++nt) {
      v16h b0 = load_bfrag(wp, FB_E4 + (ex * 2 + 0) * 2 + nt, lane);
      v16h b1 = load_bfrag(wp, FB_E4 + (ex * 2 + 1) * 2 + nt, lane);
      float bv = be4[ex * NX + nt * 16 + n];
#pragma unroll
      for (int mt = 0; mt < 2; ++mt) {
        v8f c = {};
        c = wmma_f16(ah[mt][0], b0, c);
        c = wmma_f16(ah[mt][1], b1, c);
#pragma unroll
        for (int r = 0; r < 8; ++r) {
          if (mode_r[mt][r] == ex)
            out[(tokBase + mt * 16 + r + 8 * hi) * NX + nt * 16 + n] =
                c[r] + bv;
        }
      }
    }
  }
}

// ---------------------------------------------------------------------------
// Launch
// ---------------------------------------------------------------------------
extern "C" void kernel_launch(void* const* d_in, const int* in_sizes, int n_in,
                              void* d_out, int out_size, void* d_ws,
                              size_t ws_size, hipStream_t stream) {
  const float* obs = (const float*)d_in[0];
  const float* Wc1 = (const float*)d_in[1];
  const float* bc1 = (const float*)d_in[2];
  const float* Wc2 = (const float*)d_in[3];
  const float* bc2 = (const float*)d_in[4];
  const float* Wc3 = (const float*)d_in[5];
  const float* bc3 = (const float*)d_in[6];
  const float* Wc4 = (const float*)d_in[7];
  const float* bc4 = (const float*)d_in[8];
  const float* We1 = (const float*)d_in[9];
  const float* be1 = (const float*)d_in[10];
  const float* We2 = (const float*)d_in[11];
  const float* be2 = (const float*)d_in[12];
  const float* We3 = (const float*)d_in[13];
  const float* be3 = (const float*)d_in[14];
  const float* We4 = (const float*)d_in[15];
  const float* be4 = (const float*)d_in[16];
  float* out = (float*)d_out;

  _Float16* wp = (_Float16*)d_ws;

  // Pack all weight matrices into f16 B-fragments (cheap; weights ~0.7 MB).
  pack_frags<<<20, 512, 0, stream>>>(Wc1, wp + (size_t)FB_C1 * 512, IN_DIM, HID, 5, 4);
  pack_frags<<<8, 512, 0, stream>>>(Wc2, wp + (size_t)FB_C2 * 512, HID, HID, 2, 4);
  pack_frags<<<8, 512, 0, stream>>>(Wc3, wp + (size_t)FB_C3 * 512, HID, HID, 2, 4);
  pack_frags<<<2, 512, 0, stream>>>(Wc4, wp + (size_t)FB_C4 * 512, HID, NMODES, 2, 1);
  pack_frags<<<160, 512, 0, stream>>>(We1, wp + (size_t)FB_E1 * 512, IN_DIM, HID, 5, 4);
  pack_frags<<<64, 512, 0, stream>>>(We2, wp + (size_t)FB_E2 * 512, HID, HID, 2, 4);
  pack_frags<<<64, 512, 0, stream>>>(We3, wp + (size_t)FB_E3 * 512, HID, HID, 2, 4);
  pack_frags<<<32, 512, 0, stream>>>(We4, wp + (size_t)FB_E4 * 512, HID, NX, 2, 2);

  int B = in_sizes[0] / IN_DIM;     // 131072
  int blocks = B / 128;             // 128 tokens per block (4 waves x 32)
  moe_kernel<<<blocks, 128, 0, stream>>>(obs, wp, bc1, bc2, bc3, bc4,
                                         be1, be2, be3, be4, out);
}
